// IAGNN_46359876993586
// MI455X (gfx1250) — matrix operand order repaired and checked by reading, not compiled, not run on hardware
//
#include <hip/hip_runtime.h>
#include <hip/hip_bf16.h>

// ---------------------------------------------------------------------------
// IAGNN session-graph attention for MI455X (gfx1250, wave32, WMMA).
//   out[b] = [ segment_softmax-weighted sum of ifeat | segment mean of ifeat ]
// Heavy GEMMs (ifeat@Wu^T, anchor@Wv^T) run on v_wmma_f32_16x16x32_f16
// (f16 inputs, f32 accumulate); everything feeding the final outputs stays f32.
// ---------------------------------------------------------------------------

typedef __attribute__((ext_vector_type(16))) _Float16 v16h;
typedef __attribute__((ext_vector_type(8)))  _Float16 v8h;
typedef __attribute__((ext_vector_type(8)))  float    v8f;
typedef __attribute__((ext_vector_type(4)))  float    v4f;

#define DDIM 256   // item_dim == output_dim

// first index i in [0,n) with a[i] >= key (a ascending)
__device__ __forceinline__ int lower_bound_i32(const int* __restrict__ a, int n, int key) {
  int lo = 0, hi = n;
  while (lo < hi) { int mid = (lo + hi) >> 1; if (a[mid] < key) lo = mid + 1; else hi = mid; }
  return lo;
}

// A fragment (16x32, f16) per CDNA5 ISA "16-bit A-Matrix 16x32":
// lane l row m=l%16; halves 0..7 -> K = kk+8*(l/16)+0..7, halves 8..15 -> +16.
__device__ __forceinline__ v16h load_a_frag_f32(const float* __restrict__ A, int row0, int lane, int kk) {
  const int m    = row0 + (lane & 15);
  const int koff = kk + ((lane >> 4) << 3);
  const float* p = A + m * DDIM + koff;
  v4f f0 = *reinterpret_cast<const v4f*>(p);
  v4f f1 = *reinterpret_cast<const v4f*>(p + 4);
  v4f f2 = *reinterpret_cast<const v4f*>(p + 16);
  v4f f3 = *reinterpret_cast<const v4f*>(p + 20);
  v16h a;
#pragma unroll
  for (int i = 0; i < 4; ++i) {
    a[i]      = (_Float16)f0[i];
    a[4 + i]  = (_Float16)f1[i];
    a[8 + i]  = (_Float16)f2[i];
    a[12 + i] = (_Float16)f3[i];
  }
  return a;
}

// B fragment (32x16, f16), B = W^T so B[k][n] = W[n][k] (W row-major f16):
// lane l col n=l%16; halves 0..7 -> K = kk+8*(l/16)+0..7, halves 8..15 -> +16.
__device__ __forceinline__ v16h load_b_frag(const _Float16* __restrict__ Wh, int ncol0, int lane, int kk) {
  const int n  = ncol0 + (lane & 15);
  const int kb = kk + ((lane >> 4) << 3);
  const _Float16* p = Wh + n * DDIM + kb;
  v8h b0 = *reinterpret_cast<const v8h*>(p);
  v8h b1 = *reinterpret_cast<const v8h*>(p + 16);
  v16h b;
#pragma unroll
  for (int i = 0; i < 8; ++i) { b[i] = b0[i]; b[8 + i] = b1[i]; }
  return b;
}

// --------------------------- K0: weight f32 -> f16 -------------------------
__global__ __launch_bounds__(256) void cvt_weights_kernel(
    const float* __restrict__ Wu, const float* __restrict__ Wv,
    _Float16* __restrict__ Wuh, _Float16* __restrict__ Wvh) {
  int i = blockIdx.x * 256 + threadIdx.x;
  if (i < DDIM * DDIM) {
    Wuh[i] = (_Float16)Wu[i];
    Wvh[i] = (_Float16)Wv[i];
  }
}

// --------------------------- K1: segment mean (anchor) ---------------------
__global__ __launch_bounds__(256) void anchor_kernel(
    const float* __restrict__ ifeat, const int* __restrict__ seg,
    float* __restrict__ anchor, float* __restrict__ out, int N) {
  const int b = blockIdx.x;
  __shared__ int se[2];
  if (threadIdx.x == 0) {
    se[0] = lower_bound_i32(seg, N, b);
    se[1] = lower_bound_i32(seg, N, b + 1);
  }
  __syncthreads();
  const int s = se[0], e = se[1];
  const int d = threadIdx.x;
  float sum = 0.f;
  for (int n = s; n < e; ++n) sum += ifeat[n * DDIM + d];
  const float a = sum / fmaxf((float)(e - s), 1.f);
  anchor[b * DDIM + d] = a;
  out[b * (2 * DDIM) + DDIM + d] = a;   // right half of output row = anchor
}

// ---------------- K2: feat_v = anchor @ Wv^T + bv  (WMMA) ------------------
// 256 threads = 8 waves; block handles 16 rows x 256 cols; wave owns 2 col-tiles.
__global__ __launch_bounds__(256) void featv_kernel(
    const float* __restrict__ anchor, const _Float16* __restrict__ Wvh,
    const float* __restrict__ bv, float* __restrict__ featv) {
  const int row0 = blockIdx.x * 16;
  const int lane = threadIdx.x & 31;
  const int wv   = threadIdx.x >> 5;
  const int n0 = 32 * wv, n1 = 32 * wv + 16;
  v8f acc0 = {}; v8f acc1 = {};
#pragma unroll
  for (int kk = 0; kk < DDIM; kk += 32) {
    v16h a  = load_a_frag_f32(anchor, row0, lane, kk);
    v16h b0 = load_b_frag(Wvh, n0, lane, kk);
    acc0 = __builtin_amdgcn_wmma_f32_16x16x32_f16(false, a, false, b0, (short)0, acc0, false, false);
    v16h b1 = load_b_frag(Wvh, n1, lane, kk);
    acc1 = __builtin_amdgcn_wmma_f32_16x16x32_f16(false, a, false, b1, (short)0, acc1, false, false);
  }
  // C layout: vgpr v -> row m = v + 8*(lane/16), col = l%16 within tile
  const int colA = n0 + (lane & 15);
  const int colB = n1 + (lane & 15);
  const float bvA = bv[colA], bvB = bv[colB];
#pragma unroll
  for (int v = 0; v < 8; ++v) {
    const int r = row0 + v + ((lane >> 4) << 3);
    featv[r * DDIM + colA] = acc0[v] + bvA;
    featv[r * DDIM + colB] = acc1[v] + bvB;
  }
}

// ------- K3: e[n] = sum_c sigmoid((ifeat@Wu^T)[n,c] + feat_v[seg,c])*we[c] -
// WMMA GEMM fused with sigmoid/we epilogue; [N,256] intermediate never stored.
__global__ __launch_bounds__(256) void logits_kernel(
    const float* __restrict__ ifeat, const _Float16* __restrict__ Wuh,
    const float* __restrict__ featv, const float* __restrict__ we,
    const int* __restrict__ seg, float* __restrict__ e_out) {
  __shared__ float part[8 * 16];
  const int row0 = blockIdx.x * 16;
  const int lane = threadIdx.x & 31;
  const int wv   = threadIdx.x >> 5;
  const int n0 = 32 * wv, n1 = 32 * wv + 16;
  v8f acc0 = {}; v8f acc1 = {};
#pragma unroll
  for (int kk = 0; kk < DDIM; kk += 32) {
    v16h a  = load_a_frag_f32(ifeat, row0, lane, kk);
    v16h b0 = load_b_frag(Wuh, n0, lane, kk);
    acc0 = __builtin_amdgcn_wmma_f32_16x16x32_f16(false, a, false, b0, (short)0, acc0, false, false);
    v16h b1 = load_b_frag(Wuh, n1, lane, kk);
    acc1 = __builtin_amdgcn_wmma_f32_16x16x32_f16(false, a, false, b1, (short)0, acc1, false, false);
  }
  const int colA = n0 + (lane & 15);
  const int colB = n1 + (lane & 15);
  const float weA = we[colA], weB = we[colB];
#pragma unroll
  for (int v = 0; v < 8; ++v) {
    const int m    = v + ((lane >> 4) << 3);
    const int node = row0 + m;
    const int sg   = seg[node];
    const float xA = acc0[v] + featv[sg * DDIM + colA];
    const float xB = acc1[v] + featv[sg * DDIM + colB];
    float s = weA / (1.f + __expf(-xA)) + weB / (1.f + __expf(-xB));
    // deterministic 16-lane tree reduction over the column dimension
#pragma unroll
    for (int off = 1; off < 16; off <<= 1) s += __shfl_xor(s, off, 32);
    if ((lane & 15) == 0) part[wv * 16 + m] = s;   // lane0 -> m=v, lane16 -> m=v+8
  }
  __syncthreads();
  if (threadIdx.x < 16) {
    float t = 0.f;
#pragma unroll
    for (int w2 = 0; w2 < 8; ++w2) t += part[w2 * 16 + threadIdx.x];
    e_out[row0 + threadIdx.x] = t;   // TAO == 1.0
  }
}

// ---------- K4: segment softmax + alpha-weighted segment sum ---------------
__global__ __launch_bounds__(256) void softmax_wsum_kernel(
    const float* __restrict__ ifeat, const int* __restrict__ seg,
    const float* __restrict__ e_arr, float* __restrict__ out, int N) {
  const int b = blockIdx.x;
  const int tid = threadIdx.x;
  __shared__ int se[2];
  __shared__ float red[256];
  __shared__ float wts[256];
  if (tid == 0) {
    se[0] = lower_bound_i32(seg, N, b);
    se[1] = lower_bound_i32(seg, N, b + 1);
  }
  __syncthreads();
  const int s = se[0], e = se[1];
  if (s == e) { out[b * (2 * DDIM) + tid] = 0.f; return; }  // uniform per block

  // pass 1: segment max
  float m = -INFINITY;
  for (int n = s + tid; n < e; n += 256) m = fmaxf(m, e_arr[n]);
  red[tid] = m; __syncthreads();
  for (int off = 128; off > 0; off >>= 1) {
    if (tid < off) red[tid] = fmaxf(red[tid], red[tid + off]);
    __syncthreads();
  }
  const float emax = red[0];
  __syncthreads();

  // pass 2: denom
  float sum = 0.f;
  for (int n = s + tid; n < e; n += 256) sum += __expf(e_arr[n] - emax);
  red[tid] = sum; __syncthreads();
  for (int off = 128; off > 0; off >>= 1) {
    if (tid < off) red[tid] += red[tid + off];
    __syncthreads();
  }
  const float inv = 1.f / red[0];
  __syncthreads();

  // pass 3: chunked weighted sum; thread tid owns output dim tid
  float acc = 0.f;
  for (int base = s; base < e; base += 256) {
    const int i = base + tid;
    wts[tid] = (i < e) ? __expf(e_arr[i] - emax) * inv : 0.f;
    __syncthreads();
    const int lim = (e - base < 256) ? (e - base) : 256;
    for (int j = 0; j < lim; ++j) acc += ifeat[(base + j) * DDIM + tid] * wts[j];
    __syncthreads();
  }
  out[b * (2 * DDIM) + tid] = acc;   // left half of output row = rst
}

// ---------------------------------------------------------------------------
extern "C" void kernel_launch(void* const* d_in, const int* in_sizes, int n_in,
                              void* d_out, int out_size, void* d_ws, size_t ws_size,
                              hipStream_t stream) {
  // setup_inputs() dict order: ifeat, Wu, Wv, bv, we, seg_ids
  const float* ifeat = (const float*)d_in[0];
  const float* Wu    = (const float*)d_in[1];
  const float* Wv    = (const float*)d_in[2];
  const float* bv    = (const float*)d_in[3];
  const float* we    = (const float*)d_in[4];
  const int*   seg   = (const int*)d_in[5];
  float* out = (float*)d_out;

  const int N = in_sizes[0] / DDIM;        // 204800 (multiple of 16)
  const int B = out_size / (2 * DDIM);     // 4096

  // workspace layout (bytes): Wuh | Wvh | anchor | feat_v | e
  char* ws = (char*)d_ws;
  size_t off = 0;
  _Float16* Wuh    = (_Float16*)(ws + off); off += (size_t)DDIM * DDIM * sizeof(_Float16);
  _Float16* Wvh    = (_Float16*)(ws + off); off += (size_t)DDIM * DDIM * sizeof(_Float16);
  float*    anchor = (float*)(ws + off);    off += (size_t)B * DDIM * sizeof(float);
  float*    featv  = (float*)(ws + off);    off += (size_t)B * DDIM * sizeof(float);
  float*    e_arr  = (float*)(ws + off);    off += (size_t)N * sizeof(float);
  (void)ws_size; (void)n_in;

  cvt_weights_kernel<<<(DDIM * DDIM + 255) / 256, 256, 0, stream>>>(Wu, Wv, Wuh, Wvh);
  anchor_kernel<<<B, 256, 0, stream>>>(ifeat, seg, anchor, out, N);
  featv_kernel<<<B / 16, 256, 0, stream>>>(anchor, Wvh, bv, featv);
  logits_kernel<<<N / 16, 256, 0, stream>>>(ifeat, Wuh, featv, we, seg, e_arr);
  softmax_wsum_kernel<<<B, 256, 0, stream>>>(ifeat, seg, e_arr, out, N);
}
